// DiffPool_88734024335363
// MI455X (gfx1250) — matrix operand order repaired and checked by reading, not compiled, not run on hardware
//
#include <hip/hip_runtime.h>

typedef unsigned short u16;
typedef __attribute__((ext_vector_type(16))) __bf16 v16bf;
typedef __attribute__((ext_vector_type(8)))  float  v8f;
typedef __attribute__((ext_vector_type(8)))  u16    v8u16;   // 16 bytes
typedef __attribute__((ext_vector_type(16))) u16    v16u16;  // 32 bytes

__device__ __forceinline__ u16 f32_to_bf16(float f) {
    unsigned u = __float_as_uint(f);
    unsigned r = u + 0x7FFFu + ((u >> 16) & 1u);   // round-to-nearest-even
    return (u16)(r >> 16);
}
__device__ __forceinline__ float bf16_to_f32(u16 h) {
    return __uint_as_float(((unsigned)h) << 16);
}

// gfx1250 async global->LDS copy (16B per lane), tracked by ASYNCcnt.
// GV mode: 64-bit global address per lane; dest operand is the LDS byte address.
__device__ __forceinline__ void async_copy_b128(const u16* gsrc, u16* lds_dst) {
    asm volatile("global_load_async_to_lds_b128 %0, %1, off"
                 :: "v"((unsigned)(unsigned long long)(uintptr_t)lds_dst),
                    "v"((unsigned long long)(uintptr_t)gsrc)
                 : "memory");
}
__device__ __forceinline__ void wait_async0() {
    asm volatile("s_wait_asynccnt 0x0" ::: "memory");
}

// ---------------------------------------------------------------------------
// fp32 -> bf16 conversion (grid-stride)
// ---------------------------------------------------------------------------
__global__ void cvt_f32_bf16(const float* __restrict__ in, u16* __restrict__ out, int n) {
    int i = blockIdx.x * blockDim.x + threadIdx.x;
    int stride = gridDim.x * blockDim.x;
    for (; i < n; i += stride) out[i] = f32_to_bf16(in[i]);
}

// ---------------------------------------------------------------------------
// row sums of a bf16 [rows x N] matrix -> fp32 sums
// ---------------------------------------------------------------------------
__global__ void rowsum_bf16(const u16* __restrict__ a, float* __restrict__ sums, int N) {
    __shared__ float red[256];
    const int row = blockIdx.x;
    const u16* p = a + (size_t)row * N;
    float acc = 0.f;
    for (int i = threadIdx.x; i < N; i += 256) acc += bf16_to_f32(p[i]);
    red[threadIdx.x] = acc;
    __syncthreads();
    for (int s = 128; s > 0; s >>= 1) {
        if (threadIdx.x < s) red[threadIdx.x] += red[threadIdx.x + s];
        __syncthreads();
    }
    if (threadIdx.x == 0) sums[row] = red[0];
}

// ---------------------------------------------------------------------------
// Tiled bf16 WMMA GEMM: C(MxN) = A(MxK) * B(KxN), fp32 accumulate.
// Block tile 128x128, 8 waves (wave32) as 4(M) x 2(N); each wave 32x64
// (2x4 WMMA tiles), K-step 32 via v_wmma_f32_16x16x32_bf16.
// Double-buffered LDS: A tile prefetched via global_load_async_to_lds_b128
// (ASYNCcnt), B tile prefetched to regs and stored transposed after compute.
// Epilogue MODE: 0: relu(acc+bias[n]) -> bf16
//               1: exp(acc+bias[n]) * adj[m,n] -> bf16   (unnormalized scores)
//               2: acc / sums[m] -> fp32                 (row-normalized a@s)
// ---------------------------------------------------------------------------
template <int MODE>
__launch_bounds__(256)
__global__ void gemm_bf16_wmma(const u16* __restrict__ A, const u16* __restrict__ B,
                               u16* __restrict__ outH, float* __restrict__ outF,
                               const float* __restrict__ bias,
                               const float* __restrict__ adj,
                               const float* __restrict__ sums,
                               int M, int Nn, int K) {
    constexpr int LS = 40;  // 32 + 8 halves pad: 80B rows = 20 banks (odd*4) -> conflict-free b128
    __shared__ alignas(16) u16 As[2][128 * LS];   // A tiles, row-major [m][k]
    __shared__ alignas(16) u16 Bt[2][128 * LS];   // B tiles, transposed [n][k]

    const int tid   = threadIdx.x;
    const int lane  = tid & 31;
    const int wid   = tid >> 5;          // 0..7
    const int waveM = wid >> 1;          // 0..3 (32 rows each)
    const int waveN = wid & 1;           // 0..1 (64 cols each)
    const int l16   = lane & 15;
    const int half  = lane >> 4;

    const int mbase = blockIdx.y * 128;
    const int nbase = blockIdx.x * 128;

    v8f acc[2][4];
    v8f zero = {};
    #pragma unroll
    for (int i = 0; i < 2; ++i)
        #pragma unroll
        for (int j = 0; j < 4; ++j) acc[i][j] = zero;

    // staging index precompute (per-thread constants)
    const int arow = tid >> 2, aseg = tid & 3;    // A: 128 rows x 4 x 16B chunks
    const int bk0  = tid >> 4, bseg = tid & 15;   // B: 32 rows(K) x 16 x 16B chunks

    auto stageA_async = [&](int kt, int p) {
        #pragma unroll
        for (int it = 0; it < 2; ++it) {
            int row = arow + it * 64;
            async_copy_b128(A + (size_t)(mbase + row) * K + kt * 32 + aseg * 8,
                            &As[p][row * LS + aseg * 8]);
        }
    };
    auto loadB_regs = [&](int kt, v8u16* r) {
        #pragma unroll
        for (int it = 0; it < 2; ++it) {
            int k = bk0 + it * 16;
            r[it] = *(const v8u16*)(B + (size_t)(kt * 32 + k) * Nn + nbase + bseg * 8);
        }
    };
    auto storeB_trans = [&](const v8u16* r, int p) {
        #pragma unroll
        for (int it = 0; it < 2; ++it) {
            int k = bk0 + it * 16;
            #pragma unroll
            for (int e = 0; e < 8; ++e)
                Bt[p][(bseg * 8 + e) * LS + k] = r[it][e];
        }
    };

    // ---- prologue: stage tile 0 into buffer 0 ------------------------------
    v8u16 breg[2];
    stageA_async(0, 0);
    loadB_regs(0, breg);
    storeB_trans(breg, 0);
    wait_async0();
    __syncthreads();

    const int ksteps = K >> 5;
    for (int kt = 0; kt < ksteps; ++kt) {
        const int p = kt & 1;
        const bool pre = (kt + 1) < ksteps;
        if (pre) {
            stageA_async(kt + 1, 1 - p);     // DMA next A tile while we compute
            loadB_regs(kt + 1, breg);        // next B tile in flight to regs
        }

        // ---- fragment loads per documented 16-bit WMMA VGPR layouts --------
        v16bf afr[2], bfr[4];
        #pragma unroll
        for (int i = 0; i < 2; ++i) {
            int m  = waveM * 32 + i * 16 + l16;
            int ka = half * 8;               // half0: K{0..7,16..23}; half1: +8
            v8u16 lo = *(const v8u16*)&As[p][m * LS + ka];
            v8u16 hi = *(const v8u16*)&As[p][m * LS + ka + 16];
            v16u16 t;
            #pragma unroll
            for (int e = 0; e < 8; ++e) { t[e] = lo[e]; t[8 + e] = hi[e]; }
            afr[i] = __builtin_bit_cast(v16bf, t);
        }
        #pragma unroll
        for (int j = 0; j < 4; ++j) {
            int n  = waveN * 64 + j * 16 + l16;
            int kb = half * 8;               // half0: K{0..7,16..23}; half1: +8
            v8u16 lo = *(const v8u16*)&Bt[p][n * LS + kb];
            v8u16 hi = *(const v8u16*)&Bt[p][n * LS + kb + 16];
            v16u16 t;
            #pragma unroll
            for (int e = 0; e < 8; ++e) { t[e] = lo[e]; t[8 + e] = hi[e]; }
            bfr[j] = __builtin_bit_cast(v16bf, t);
        }

        #pragma unroll
        for (int i = 0; i < 2; ++i)
            #pragma unroll
            for (int j = 0; j < 4; ++j)
                acc[i][j] = __builtin_amdgcn_wmma_f32_16x16x32_bf16(
                    false, afr[i], false, bfr[j], (short)0, acc[i][j], false, false);

        if (pre) {
            storeB_trans(breg, 1 - p);       // commit next B tile (other buffer)
            wait_async0();                   // next A tile fully in LDS (this wave)
        }
        __syncthreads();                     // all waves' staging visible
    }

    // ---- epilogue: C/D layout: VGPR r -> (M = r + half*8, N = l16) ---------
    const int mrow = mbase + waveM * 32;
    const int ncol = nbase + waveN * 64;
    float bcol[4] = {0.f, 0.f, 0.f, 0.f};
    if (MODE == 0 || MODE == 1) {
        #pragma unroll
        for (int j = 0; j < 4; ++j) bcol[j] = bias[ncol + j * 16 + l16];
    }
    #pragma unroll
    for (int i = 0; i < 2; ++i) {
        #pragma unroll
        for (int j = 0; j < 4; ++j) {
            #pragma unroll
            for (int r = 0; r < 8; ++r) {
                int row = mrow + i * 16 + half * 8 + r;
                int col = ncol + j * 16 + l16;
                float v = acc[i][j][r];
                if (MODE == 0) {
                    v += bcol[j];
                    v = v > 0.f ? v : 0.f;
                    outH[(size_t)row * Nn + col] = f32_to_bf16(v);
                } else if (MODE == 1) {
                    v = __expf(v + bcol[j]) * adj[(size_t)row * Nn + col];
                    outH[(size_t)row * Nn + col] = f32_to_bf16(v);
                } else {
                    outF[(size_t)row * Nn + col] = v / sums[row];
                }
            }
        }
    }
}

// ---------------------------------------------------------------------------
// Host-side orchestration
// ---------------------------------------------------------------------------
extern "C" void kernel_launch(void* const* d_in, const int* in_sizes, int n_in,
                              void* d_out, int out_size, void* d_ws, size_t ws_size,
                              hipStream_t stream) {
    constexpr int N    = 4096;
    constexpr int FIN  = 512;
    constexpr int FOUT = 4096;

    const float* x   = (const float*)d_in[0];
    const float* adj = (const float*)d_in[1];
    const float* W1  = (const float*)d_in[2];
    const float* b1  = (const float*)d_in[3];
    const float* W2  = (const float*)d_in[4];
    const float* b2  = (const float*)d_in[5];
    const float* W3  = (const float*)d_in[6];
    const float* b3  = (const float*)d_in[7];
    float* out = (float*)d_out;

    char* ws = (char*)d_ws;
    size_t off = 0;
    auto alloc = [&](size_t bytes) { char* p = ws + off; off += (bytes + 255) & ~size_t(255); return p; };
    u16*   xb   = (u16*)  alloc((size_t)N * FIN  * 2);   // x  bf16
    u16*   W1b  = (u16*)  alloc((size_t)FIN * FOUT * 2); // W1 bf16
    u16*   W2b  = (u16*)  alloc((size_t)FIN * FOUT * 2); // W2 bf16
    u16*   W3b  = (u16*)  alloc((size_t)FOUT * FOUT * 2);// W3 bf16
    u16*   hb   = (u16*)  alloc((size_t)N * FOUT * 2);   // h  bf16
    u16*   sb   = (u16*)  alloc((size_t)N * FOUT * 2);   // s  bf16
    u16*   ab   = (u16*)  alloc((size_t)N * N * 2);      // unnormalized scores bf16
    float* sums = (float*)alloc((size_t)N * 4);          // row sums fp32

    // 1) convert fp32 inputs to bf16
    cvt_f32_bf16<<<2048, 256, 0, stream>>>(x,  xb,  N * FIN);
    cvt_f32_bf16<<<2048, 256, 0, stream>>>(W1, W1b, FIN * FOUT);
    cvt_f32_bf16<<<2048, 256, 0, stream>>>(W2, W2b, FIN * FOUT);
    cvt_f32_bf16<<<2048, 256, 0, stream>>>(W3, W3b, FOUT * FOUT);

    dim3 blk(256);
    dim3 grd(N / 128, N / 128);   // 128x128 block tiles; all dims divide evenly
    // 2) h = relu(x@W1 + b1) ; s = relu(x@W2 + b2)
    gemm_bf16_wmma<0><<<grd, blk, 0, stream>>>(xb, W1b, hb, nullptr, b1, nullptr, nullptr, N, FOUT, FIN);
    gemm_bf16_wmma<0><<<grd, blk, 0, stream>>>(xb, W2b, sb, nullptr, b2, nullptr, nullptr, N, FOUT, FIN);
    // 3) a_un = exp(h@W3 + b3) * adj
    gemm_bf16_wmma<1><<<grd, blk, 0, stream>>>(hb, W3b, ab, nullptr, b3, adj, nullptr, N, N, FOUT);
    // 4) row sums of unnormalized scores
    rowsum_bf16<<<N, 256, 0, stream>>>(ab, sums, N);
    // 5) out = (a_un @ s) / rowsum[row]   (normalization folded into epilogue)
    gemm_bf16_wmma<2><<<grd, blk, 0, stream>>>(ab, sb, nullptr, out, nullptr, nullptr, sums, N, FOUT, N);
}
